// VQVAE_53128745452293
// MI455X (gfx1250) — matrix operand order repaired and checked by reading, not compiled, not run on hardware
//
#include <hip/hip_runtime.h>
#include <hip/hip_bf16.h>

typedef unsigned short u16;
typedef unsigned int   u32;
typedef unsigned long long u64;

typedef __attribute__((ext_vector_type(8)))  __bf16 v8bf;
typedef __attribute__((ext_vector_type(16))) __bf16 v16bf;
typedef __attribute__((ext_vector_type(8)))  float  v8f;

union BF16x8  { v8bf v; u16 s[8]; };
union BF16x16 { v16bf v; v8bf h[2]; };

__device__ __forceinline__ u16 f32_to_bf16(float f) {
  u32 u = __float_as_uint(f);
  u32 r = u + 0x7FFFu + ((u >> 16) & 1u);   // round-to-nearest-even
  return (u16)(r >> 16);
}
__device__ __forceinline__ float bf16_to_f32(u16 h) {
  return __uint_as_float(((u32)h) << 16);
}

// ---------------------------------------------------------------------------
// fp32 -> bf16 elementwise (vectorized x4)
// ---------------------------------------------------------------------------
__global__ void k_cvt_bf16(const float* __restrict__ in, u16* __restrict__ out, int n4) {
  int i = blockIdx.x * blockDim.x + threadIdx.x;
  int stride = gridDim.x * blockDim.x;
  for (; i < n4; i += stride) {
    float4 f = ((const float4*)in)[i];
    uint2 p;
    p.x = (u32)f32_to_bf16(f.x) | ((u32)f32_to_bf16(f.y) << 16);
    p.y = (u32)f32_to_bf16(f.z) | ((u32)f32_to_bf16(f.w) << 16);
    ((uint2*)out)[i] = p;
  }
}

// ---------------------------------------------------------------------------
// row-wise sum of squares, 256 columns, one wave per row (8 rows / block)
// ---------------------------------------------------------------------------
__global__ void k_rownorm_f32(const float* __restrict__ src, float* __restrict__ out) {
  int w = threadIdx.x >> 5, lane = threadIdx.x & 31;
  int row = blockIdx.x * 8 + w;
  const float4* p = (const float4*)(src + (size_t)row * 256);
  float4 a = p[lane * 2], b = p[lane * 2 + 1];
  float s = a.x*a.x + a.y*a.y + a.z*a.z + a.w*a.w
          + b.x*b.x + b.y*b.y + b.z*b.z + b.w*b.w;
  #pragma unroll
  for (int off = 16; off >= 1; off >>= 1) s += __shfl_xor(s, off, 32);
  if (lane == 0) out[row] = s;
}

__global__ void k_rownorm_bf16(const u16* __restrict__ src, float* __restrict__ out) {
  int w = threadIdx.x >> 5, lane = threadIdx.x & 31;
  int row = blockIdx.x * 8 + w;
  BF16x8 e; e.v = *(const v8bf*)(src + (size_t)row * 256 + lane * 8);
  float s = 0.0f;
  #pragma unroll
  for (int j = 0; j < 8; ++j) { float f = bf16_to_f32(e.s[j]); s += f * f; }
  #pragma unroll
  for (int off = 16; off >= 1; off >>= 1) s += __shfl_xor(s, off, 32);
  if (lane == 0) out[row] = s;
}

// ---------------------------------------------------------------------------
// Generic bf16 WMMA GEMM: C = [relu](A @ B + bias), A:[M,K], B:[K,N] row-major
// Block tile 128x128, K-step 32; 8 waves as 4x2, wave tile 32x64 (2x4 frags).
// Straight-line LDS fills with hoisted, stride-advanced pointers.
// ---------------------------------------------------------------------------
#define BM  128
#define BN  128
#define BKg 32
#define BKP 40   // padded LDS row stride (halfs): 80B keeps 16B alignment, spreads banks

__global__ __launch_bounds__(256) void k_gemm_bf16(
    const u16* __restrict__ A, const u16* __restrict__ B,
    const float* __restrict__ bias,
    u16* __restrict__ outBf, float* __restrict__ outF,
    int Ncols, int Kdim, int relu)
{
  __shared__ u16 As[BM * BKP];   // A tile, row-major [BM][BKP]
  __shared__ u16 Bt[BN * BKP];   // B tile transposed: row n = column of B

  const int tid = threadIdx.x;
  const int m0 = blockIdx.x * BM, n0 = blockIdx.y * BN;
  const int w  = tid >> 5, lane = tid & 31;
  const int wm = w >> 1,  wn   = w & 1;
  const int lh = lane >> 4, lm = lane & 15;

  // --- hoisted fill addressing ---
  // A: 128 rows x 32 halfs = 512 x 16B chunks, 2 per thread
  const int ai  = tid >> 2;            // 0..63
  const int akc = (tid & 3) << 3;      // 0,8,16,24
  const u16* pA0 = A + (size_t)(m0 + ai) * Kdim + akc;
  const u16* pA1 = A + (size_t)(m0 + ai + 64) * Kdim + akc;
  u16* sA0 = &As[ai * BKP + akc];
  u16* sA1 = &As[(ai + 64) * BKP + akc];
  // B: 32 k-rows x 128 cols = 512 x 16B chunks, 2 per thread (transpose scatter)
  const int bkk = tid >> 4;            // 0..15 (second chunk: +16)
  const int bnc = (tid & 15) << 3;     // 0..120
  const u16* pB0 = B + (size_t)bkk * Ncols + n0 + bnc;
  const u16* pB1 = pB0 + (size_t)16 * Ncols;
  // --- hoisted fragment pointers (loop-invariant LDS addresses) ---
  const u16* paF[2]; const u16* pbF[4];
  #pragma unroll
  for (int fm = 0; fm < 2; ++fm) paF[fm] = &As[(wm * 32 + fm * 16 + lm) * BKP + lh * 8];
  #pragma unroll
  for (int fn = 0; fn < 4; ++fn) pbF[fn] = &Bt[(wn * 64 + fn * 16 + lm) * BKP + lh * 8];

  v8f acc[2][4] = {};
  const size_t strideB = (size_t)BKg * Ncols;

  for (int k0 = 0; k0 < Kdim; k0 += BKg) {
    v8bf a0 = *(const v8bf*)pA0; pA0 += BKg;
    v8bf a1 = *(const v8bf*)pA1; pA1 += BKg;
    BF16x8 b0; b0.v = *(const v8bf*)pB0; pB0 += strideB;
    BF16x8 b1; b1.v = *(const v8bf*)pB1; pB1 += strideB;
    *(v8bf*)sA0 = a0;
    *(v8bf*)sA1 = a1;
    #pragma unroll
    for (int j = 0; j < 8; ++j) {
      Bt[(bnc + j) * BKP + bkk]      = b0.s[j];
      Bt[(bnc + j) * BKP + bkk + 16] = b1.s[j];
    }
    __syncthreads();

    v16bf aF[2];
    #pragma unroll
    for (int fm = 0; fm < 2; ++fm) {
      BF16x16 t; t.h[0] = *(const v8bf*)paF[fm]; t.h[1] = *(const v8bf*)(paF[fm] + 16);
      aF[fm] = t.v;
    }
    #pragma unroll
    for (int fn = 0; fn < 4; ++fn) {
      BF16x16 t; t.h[0] = *(const v8bf*)pbF[fn]; t.h[1] = *(const v8bf*)(pbF[fn] + 16);
      #pragma unroll
      for (int fm = 0; fm < 2; ++fm)
        acc[fm][fn] = __builtin_amdgcn_wmma_f32_16x16x32_bf16(
            false, aF[fm], false, t.v, (short)0, acc[fm][fn], false, false);
    }
    __syncthreads();
  }

  // epilogue: C/D layout: VGPR v, lanes 0-15 -> row v, lanes 16-31 -> row v+8
  #pragma unroll
  for (int fm = 0; fm < 2; ++fm)
    #pragma unroll
    for (int fn = 0; fn < 4; ++fn) {
      int n = n0 + wn * 64 + fn * 16 + lm;
      float bv = bias[n];
      #pragma unroll
      for (int v = 0; v < 8; ++v) {
        int m = m0 + wm * 32 + fm * 16 + lh * 8 + v;
        float val = acc[fm][fn][v] + bv;
        if (relu) val = fmaxf(val, 0.0f);
        if (outF)  outF[(size_t)m * Ncols + n] = val;
        if (outBf) outBf[(size_t)m * Ncols + n] = f32_to_bf16(val);
      }
    }
}

// ---------------------------------------------------------------------------
// VQ argmin. 4 waves/block, wave owns 16 rows; A-fragments (16 rows x 256
// dims bf16) held fully in registers. Codeword tiles (32 x 256) streamed
// through double-buffered LDS with GLOBAL_LOAD_ASYNC_TO_LDS_B128 +
// s_wait_asynccnt software pipeline. Running min of (c2[k] - 2*dot).
// ---------------------------------------------------------------------------
#define EP 264   // padded row stride (halfs) for 256-dim rows

__global__ __launch_bounds__(128) void k_vq_argmin(
    const u16* __restrict__ encbf, const u16* __restrict__ cwbf,
    const float* __restrict__ c2, const float* __restrict__ x2,
    float* __restrict__ outIdxF, float* __restrict__ outDist,
    int* __restrict__ outIdxI, int Krows)
{
  __shared__ u16 Ct[2][32 * EP];   // double-buffered codeword tile

  const int tid = threadIdx.x;
  const int r0  = blockIdx.x * 64;
  const int w   = tid >> 5, lane = tid & 31;
  const int lh  = lane >> 4, lm  = lane & 15;

  // A fragments in registers: lane holds row r0+w*16+lm; per 32-dim tile d:
  // elems 0..7 at d*32 + 8*lh, elems 8..15 at d*32 + 16 + 8*lh (ISA layout).
  v16bf aF[8];
  {
    const u16* rowp = encbf + (size_t)(r0 + w * 16 + lm) * 256 + lh * 8;
    #pragma unroll
    for (int d = 0; d < 8; ++d) {
      BF16x16 t;
      t.h[0] = *(const v8bf*)(rowp + d * 32);
      t.h[1] = *(const v8bf*)(rowp + d * 32 + 16);
      aF[d] = t.v;
    }
  }

  // async fill addressing: 32 rows x 512B = 1024 x 16B chunks, 8 per thread,
  // thread's chunk j covers codeword row kkb+4j at dim-offset chh.
  const u32 chh = (tid & 31) << 3;
  const u32 kkb = tid >> 5;
  const u16* gbase = cwbf + (size_t)kkb * 256 + chh;
  u32 ldsC[2];
  ldsC[0] = (u32)(size_t)&Ct[0][kkb * EP + chh];
  ldsC[1] = (u32)(size_t)&Ct[1][kkb * EP + chh];

  const int ntiles = Krows / 32;   // 256

  // prologue: issue tile 0 into buffer 0
  #pragma unroll
  for (int j = 0; j < 8; ++j) {
    u32 l = ldsC[0] + (u32)(4 * j * EP * 2);
    u64 g = (u64)(size_t)(gbase + (size_t)4 * j * 256);
    asm volatile("global_load_async_to_lds_b128 %0, %1, off"
                 :: "v"(l), "v"(g) : "memory");
  }

  float best[8]; int bidx[8];
  #pragma unroll
  for (int v = 0; v < 8; ++v) { best[v] = 3.4e38f; bidx[v] = 0; }

  for (int tile = 0; tile < ntiles; ++tile) {
    const int buf = tile & 1;
    if (tile + 1 < ntiles) {
      // issue next tile into the other buffer, then wait for current tile
      const u16* gnext = gbase + (size_t)(tile + 1) * 32 * 256;
      const u32 lbase = ldsC[buf ^ 1];
      #pragma unroll
      for (int j = 0; j < 8; ++j) {
        u32 l = lbase + (u32)(4 * j * EP * 2);
        u64 g = (u64)(size_t)(gnext + (size_t)4 * j * 256);
        asm volatile("global_load_async_to_lds_b128 %0, %1, off"
                     :: "v"(l), "v"(g) : "memory");
      }
      asm volatile("s_wait_asynccnt 0x8" ::: "memory");
    } else {
      asm volatile("s_wait_asynccnt 0x0" ::: "memory");
    }
    __syncthreads();   // everyone's chunks of the current tile are in LDS

    const int k0 = tile * 32;
    #pragma unroll
    for (int sub = 0; sub < 2; ++sub) {
      const u16* pb = &Ct[buf][(sub * 16 + lm) * EP + lh * 8];
      v8f acc = {};
      #pragma unroll
      for (int d = 0; d < 8; ++d) {
        BF16x16 tb;
        tb.h[0] = *(const v8bf*)(pb + d * 32);
        tb.h[1] = *(const v8bf*)(pb + d * 32 + 16);
        acc = __builtin_amdgcn_wmma_f32_16x16x32_bf16(
            false, aF[d], false, tb.v, (short)0, acc, false, false);
      }
      int kIdx = k0 + sub * 16 + lm;
      float c2v = c2[kIdx];
      #pragma unroll
      for (int v = 0; v < 8; ++v) {
        float s = fmaf(-2.0f, acc[v], c2v);
        if (s < best[v]) { best[v] = s; bidx[v] = kIdx; }
      }
    }
    __syncthreads();   // tile consumed; next iteration may overwrite this buffer
  }

  // butterfly min-reduce across the 16 lanes of each half (rows differ by half)
  #pragma unroll
  for (int v = 0; v < 8; ++v) {
    float bv = best[v]; int bi = bidx[v];
    #pragma unroll
    for (int off = 8; off >= 1; off >>= 1) {
      float ov = __shfl_xor(bv, off, 32);
      int   oi = __shfl_xor(bi, off, 32);
      if (ov < bv || (ov == bv && oi < bi)) { bv = ov; bi = oi; }
    }
    if (lm == 0) {
      int row = r0 + w * 16 + lh * 8 + v;
      float d2 = x2[row] + bv;
      outDist[row] = sqrtf(fmaxf(d2, 0.0f));
      outIdxF[row] = (float)bi;
      outIdxI[row] = bi;
    }
  }
}

// ---------------------------------------------------------------------------
// gather codewords[idx] -> quantized (fp32 out) + bf16 copy for decoder
// ---------------------------------------------------------------------------
__global__ void k_gather(const int* __restrict__ idx, const float* __restrict__ cw,
                         float* __restrict__ outQ, u16* __restrict__ qbf)
{
  int t = blockIdx.x * 256 + threadIdx.x;
  int row = t >> 6;
  int c = (t & 63) << 2;
  int k = idx[row];
  float4 v = *(const float4*)(cw + (size_t)k * 256 + c);
  *(float4*)(outQ + (size_t)row * 256 + c) = v;
  uint2 p;
  p.x = (u32)f32_to_bf16(v.x) | ((u32)f32_to_bf16(v.y) << 16);
  p.y = (u32)f32_to_bf16(v.z) | ((u32)f32_to_bf16(v.w) << 16);
  *(uint2*)(qbf + (size_t)row * 256 + c) = p;
}

// ---------------------------------------------------------------------------
extern "C" void kernel_launch(void* const* d_in, const int* in_sizes, int n_in,
                              void* d_out, int out_size, void* d_ws, size_t ws_size,
                              hipStream_t stream)
{
  const int N = 16384, IN = 1024, E = 256, K = 8192;
  const float* x      = (const float*)d_in[0];
  const float* enc_w1 = (const float*)d_in[1];
  const float* enc_b1 = (const float*)d_in[2];
  const float* enc_w2 = (const float*)d_in[3];
  const float* enc_b2 = (const float*)d_in[4];
  const float* cw     = (const float*)d_in[5];
  const float* dec_w1 = (const float*)d_in[6];
  const float* dec_b1 = (const float*)d_in[7];
  const float* dec_w2 = (const float*)d_in[8];
  const float* dec_b2 = (const float*)d_in[9];

  // outputs: indices | distances | reconstructed | quantized (all as float)
  float* outIdx  = (float*)d_out;
  float* outDist = outIdx + N;
  float* outRec  = outDist + N;
  float* outQ    = outRec + (size_t)N * IN;

  char* p = (char*)d_ws;
  auto alloc = [&](size_t bytes) -> char* {
    char* r = p; p += (bytes + 255) & ~(size_t)255; return r;
  };
  u16* x_bf   = (u16*)alloc((size_t)N * IN * 2);
  u16* w1_bf  = (u16*)alloc((size_t)IN * E * 2);
  u16* w2_bf  = (u16*)alloc((size_t)E * E * 2);
  u16* dw1_bf = (u16*)alloc((size_t)E * E * 2);
  u16* dw2_bf = (u16*)alloc((size_t)E * IN * 2);
  u16* cw_bf  = (u16*)alloc((size_t)K * E * 2);
  u16* h_bf   = (u16*)alloc((size_t)N * E * 2);
  u16* e_bf   = (u16*)alloc((size_t)N * E * 2);
  u16* q_bf   = (u16*)alloc((size_t)N * E * 2);
  u16* g_bf   = (u16*)alloc((size_t)N * E * 2);
  float* c2   = (float*)alloc((size_t)K * 4);
  float* x2   = (float*)alloc((size_t)N * 4);
  int*   idxI = (int*)alloc((size_t)N * 4);

  auto cvt = [&](const float* src, u16* dst, size_t n) {
    int n4 = (int)(n / 4);
    int blocks = (n4 + 255) / 256; if (blocks > 4096) blocks = 4096;
    k_cvt_bf16<<<blocks, 256, 0, stream>>>(src, dst, n4);
  };
  cvt(x,      x_bf,   (size_t)N * IN);
  cvt(enc_w1, w1_bf,  (size_t)IN * E);
  cvt(enc_w2, w2_bf,  (size_t)E * E);
  cvt(dec_w1, dw1_bf, (size_t)E * E);
  cvt(dec_w2, dw2_bf, (size_t)E * IN);
  cvt(cw,     cw_bf,  (size_t)K * E);

  k_rownorm_f32<<<K / 8, 256, 0, stream>>>(cw, c2);

  // encoder
  k_gemm_bf16<<<dim3(N / BM, E / BN), 256, 0, stream>>>(
      x_bf, w1_bf, enc_b1, h_bf, (float*)nullptr, E, IN, 1);
  k_gemm_bf16<<<dim3(N / BM, E / BN), 256, 0, stream>>>(
      h_bf, w2_bf, enc_b2, e_bf, (float*)nullptr, E, E, 1);

  // VQ
  k_rownorm_bf16<<<N / 8, 256, 0, stream>>>(e_bf, x2);
  k_vq_argmin<<<N / 64, 128, 0, stream>>>(
      e_bf, cw_bf, c2, x2, outIdx, outDist, idxI, K);
  k_gather<<<(N * 64) / 256, 256, 0, stream>>>(idxI, cw, outQ, q_bf);

  // decoder
  k_gemm_bf16<<<dim3(N / BM, E / BN), 256, 0, stream>>>(
      q_bf, dw1_bf, dec_b1, g_bf, (float*)nullptr, E, E, 1);
  k_gemm_bf16<<<dim3(N / BM, IN / BN), 256, 0, stream>>>(
      g_bf, dw2_bf, dec_b2, (u16*)nullptr, outRec, IN, E, 0);
}
